// Classifier_47253230191424
// MI455X (gfx1250) — matrix operand order repaired and checked by reading, not compiled, not run on hardware
//
#include <hip/hip_runtime.h>
#include <hip/hip_bf16.h>

typedef __attribute__((ext_vector_type(16))) _Float16 v16h;
typedef __attribute__((ext_vector_type(8)))  float    v8f;

#define HGAT 8      // heads
#define FGAT 16     // feats per head
#define INF_ 256    // input feats
#define HF   128    // H*F

// ---------------------------------------------------------------------------
// Prep: W [256,128] fp32 row-major -> Wt [128][256] f16 col-major (B operand
// becomes a single contiguous 32B v16h load per lane per k-step).
// ---------------------------------------------------------------------------
__global__ void prep_wt(const float* __restrict__ W, _Float16* __restrict__ Wt) {
  int idx = blockIdx.x * blockDim.x + threadIdx.x;
  if (idx >= HF * INF_) return;
  int col = idx >> 8;        // 0..127
  int k   = idx & 255;       // 0..255
  Wt[(size_t)col * INF_ + k] = (_Float16)W[(size_t)k * HF + col];
}

// ---------------------------------------------------------------------------
// Init: rst = 0, m = -inf, sbuf = 0 (re-run every launch; graph-replay safe).
// ---------------------------------------------------------------------------
__global__ void init_bufs(float* __restrict__ rst, float* __restrict__ m,
                          float* __restrict__ sbuf, int N) {
  long long idx = (long long)blockIdx.x * blockDim.x + threadIdx.x;
  if (idx < (long long)N * HF)   rst[idx]  = 0.0f;
  if (idx < (long long)N * HGAT) { m[idx] = -__builtin_inff(); sbuf[idx] = 0.0f; }
}

// ---------------------------------------------------------------------------
// feat[N,128] = x[N,256] @ W[256,128] via v_wmma_f32_16x16x32_f16.
// blockDim = 128 (4 waves); one wave -> one 16-row stripe, all 8 col tiles.
// ---------------------------------------------------------------------------
__global__ void gemm_feat(const float* __restrict__ x,
                          const _Float16* __restrict__ Wt,
                          float* __restrict__ feat, int N) {
  const int lane    = threadIdx.x & 31;
  const int wave    = threadIdx.x >> 5;
  const int tile    = blockIdx.x * 4 + wave;
  const int rowBase = tile * 16;
  if (rowBase >= N) return;                 // wave-uniform exit

  const int row16 = lane & 15;              // A: row within tile; B/C: col within tile
  const int khalf = lane >> 4;              // which K-half this lane holds

  int grow = rowBase + row16;
  if (grow >= N) grow = N - 1;              // clamp (stores guarded below)
  const float* xrow = x + (size_t)grow * INF_;

  v8f acc[8];
#pragma unroll
  for (int ct = 0; ct < 8; ++ct) acc[ct] = (v8f){0,0,0,0,0,0,0,0};

  for (int kb = 0; kb < INF_; kb += 32) {
    // A fragment (contiguous k order, same as B):
    //   lanes 0-15 : elems 0-15 = K kb+0..15
    //   lanes 16-31: elems 0-15 = K kb+16..31
    const float4 f0 = *(const float4*)(xrow + kb + khalf * 16);
    const float4 f1 = *(const float4*)(xrow + kb + khalf * 16 + 4);
    const float4 f2 = *(const float4*)(xrow + kb + khalf * 16 + 8);
    const float4 f3 = *(const float4*)(xrow + kb + khalf * 16 + 12);
    v16h a;
    a[0]=(_Float16)f0.x;  a[1]=(_Float16)f0.y;  a[2]=(_Float16)f0.z;  a[3]=(_Float16)f0.w;
    a[4]=(_Float16)f1.x;  a[5]=(_Float16)f1.y;  a[6]=(_Float16)f1.z;  a[7]=(_Float16)f1.w;
    a[8]=(_Float16)f2.x;  a[9]=(_Float16)f2.y;  a[10]=(_Float16)f2.z; a[11]=(_Float16)f2.w;
    a[12]=(_Float16)f3.x; a[13]=(_Float16)f3.y; a[14]=(_Float16)f3.z; a[15]=(_Float16)f3.w;

#pragma unroll
    for (int ct = 0; ct < 8; ++ct) {
      // B fragment (32x16): lanes 0-15 hold K kb+0..15 of col, lanes 16-31 K kb+16..31
      const int col = ct * 16 + row16;
      const v16h b = *(const v16h*)(Wt + (size_t)col * INF_ + kb + khalf * 16);
      acc[ct] = __builtin_amdgcn_wmma_f32_16x16x32_f16(
          false, a, false, b, (short)0, acc[ct], false, false);
    }
  }

  // C/D 16x16 f32 layout: lane holds col = lane&15; VGPR r -> M = r + 8*khalf
#pragma unroll
  for (int ct = 0; ct < 8; ++ct) {
#pragma unroll
    for (int r = 0; r < 8; ++r) {
      const int mrow = rowBase + khalf * 8 + r;
      if (mrow < N) feat[(size_t)mrow * HF + ct * 16 + row16] = acc[ct][r];
    }
  }
}

// ---------------------------------------------------------------------------
// el/er [N,H]: per-head dot of feat with attn vectors.
// ---------------------------------------------------------------------------
__global__ void node_attn(const float* __restrict__ feat,
                          const float* __restrict__ attn_l,
                          const float* __restrict__ attn_r,
                          float* __restrict__ el, float* __restrict__ er, int N) {
  long long idx = (long long)blockIdx.x * blockDim.x + threadIdx.x;
  if (idx >= (long long)N * HGAT) return;
  const int h = (int)(idx & 7);
  const float* fp = feat + idx * FGAT;   // n*128 + h*16 == idx*16
  float aL = 0.f, aR = 0.f;
#pragma unroll
  for (int f = 0; f < FGAT; ++f) {
    const float v = fp[f];
    aL += v * attn_l[h * FGAT + f];
    aR += v * attn_r[h * FGAT + f];
  }
  el[idx] = aL;
  er[idx] = aR;
}

__device__ __forceinline__ float lrelu(float v) { return v > 0.f ? v : 0.2f * v; }

__device__ __forceinline__ void atomicMaxF(float* addr, float v) {
  if (v >= 0.f) atomicMax((int*)addr, __float_as_int(v));
  else          atomicMin((unsigned int*)addr, __float_as_uint(v));
}

// Pass 1: segment max of leaky_relu(el[src]+er[dst]) into m[dst,h]
__global__ void edge_max(const int* __restrict__ src, const int* __restrict__ dst,
                         const float* __restrict__ el, const float* __restrict__ er,
                         float* __restrict__ m, long long EH) {
  long long idx = (long long)blockIdx.x * blockDim.x + threadIdx.x;
  if (idx >= EH) return;
  const long long e = idx >> 3; const int h = (int)(idx & 7);
  const long long s_ = (long long)src[e], d_ = (long long)dst[e];
  const float v = lrelu(el[s_ * HGAT + h] + er[d_ * HGAT + h]);
  atomicMaxF(&m[d_ * HGAT + h], v);
}

// Pass 2: segment sum of exp(e - m[dst]) into sbuf[dst,h]
__global__ void edge_sum(const int* __restrict__ src, const int* __restrict__ dst,
                         const float* __restrict__ el, const float* __restrict__ er,
                         const float* __restrict__ m, float* __restrict__ sbuf,
                         long long EH) {
  long long idx = (long long)blockIdx.x * blockDim.x + threadIdx.x;
  if (idx >= EH) return;
  const long long e = idx >> 3; const int h = (int)(idx & 7);
  const long long s_ = (long long)src[e], d_ = (long long)dst[e];
  const float v  = lrelu(el[s_ * HGAT + h] + er[d_ * HGAT + h]);
  const float ex = __expf(v - m[d_ * HGAT + h]);
  atomicAdd(&sbuf[d_ * HGAT + h], ex);
}

// Pass 3: rst[dst,h,f] += a(e,h) * feat[src,h,f].
// 16 consecutive lanes share (e,h): broadcast scalar loads, coalesced gather+atomics.
__global__ void edge_agg(const int* __restrict__ src, const int* __restrict__ dst,
                         const float* __restrict__ el, const float* __restrict__ er,
                         const float* __restrict__ m, const float* __restrict__ sbuf,
                         const float* __restrict__ feat, float* __restrict__ rst,
                         long long EHF) {
  long long idx = (long long)blockIdx.x * blockDim.x + threadIdx.x;
  if (idx >= EHF) return;
  const long long e = idx >> 7;          // /128
  const int r = (int)(idx & 127);        // h*16 + f
  const int h = r >> 4;
  const long long s_ = (long long)src[e], d_ = (long long)dst[e];
  const float v  = lrelu(el[s_ * HGAT + h] + er[d_ * HGAT + h]);
  const float ex = __expf(v - m[d_ * HGAT + h]);
  const float a  = ex / fmaxf(sbuf[d_ * HGAT + h], 1e-9f);
  atomicAdd(&rst[d_ * HF + r], a * feat[s_ * HF + r]);
}

// Final: out[n,f] = mean_h(rst[n,h,f] + bias[h,f])
__global__ void finalize(const float* __restrict__ rst, const float* __restrict__ bias,
                         float* __restrict__ out, int N) {
  long long idx = (long long)blockIdx.x * blockDim.x + threadIdx.x;
  if (idx >= (long long)N * FGAT) return;
  const long long n = idx >> 4;
  const int f = (int)(idx & 15);
  float s = 0.f;
#pragma unroll
  for (int h = 0; h < HGAT; ++h)
    s += rst[n * HF + h * FGAT + f] + bias[h * FGAT + f];
  out[idx] = s * 0.125f;
}

extern "C" void kernel_launch(void* const* d_in, const int* in_sizes, int n_in,
                              void* d_out, int out_size, void* d_ws, size_t ws_size,
                              hipStream_t stream) {
  const float* x      = (const float*)d_in[0];
  const float* W      = (const float*)d_in[1];
  const float* attn_l = (const float*)d_in[2];
  const float* attn_r = (const float*)d_in[3];
  const float* bias   = (const float*)d_in[4];
  const int*   src    = (const int*)d_in[5];
  const int*   dst    = (const int*)d_in[6];
  float* out = (float*)d_out;

  const int N = in_sizes[0] / INF_;
  const long long E = (long long)in_sizes[5];

  // workspace layout
  char* ws = (char*)d_ws;
  size_t off = 0;
  _Float16* Wt = (_Float16*)(ws + off); off += (size_t)HF * INF_ * sizeof(_Float16); // 64 KB
  float* feat  = (float*)(ws + off);    off += (size_t)N * HF * sizeof(float);
  float* el    = (float*)(ws + off);    off += (size_t)N * HGAT * sizeof(float);
  float* er    = (float*)(ws + off);    off += (size_t)N * HGAT * sizeof(float);
  float* m     = (float*)(ws + off);    off += (size_t)N * HGAT * sizeof(float);
  float* sbuf  = (float*)(ws + off);    off += (size_t)N * HGAT * sizeof(float);
  float* rst   = (float*)(ws + off);    off += (size_t)N * HF * sizeof(float);
  (void)ws_size; (void)n_in; (void)out_size;

  const int B = 256;

  prep_wt<<<(HF * INF_ + B - 1) / B, B, 0, stream>>>(W, Wt);

  {
    long long n = (long long)N * HF;
    init_bufs<<<(int)((n + B - 1) / B), B, 0, stream>>>(rst, m, sbuf, N);
  }

  {
    const int nTiles = (N + 15) / 16;
    const int blocks = (nTiles + 3) / 4;          // 4 waves (128 thr) per block
    gemm_feat<<<blocks, 128, 0, stream>>>(x, Wt, feat, N);
  }

  {
    long long n = (long long)N * HGAT;
    node_attn<<<(int)((n + B - 1) / B), B, 0, stream>>>(feat, attn_l, attn_r, el, er, N);
  }

  {
    long long EH = E * HGAT;
    int blocks = (int)((EH + B - 1) / B);
    edge_max<<<blocks, B, 0, stream>>>(src, dst, el, er, m, EH);
    edge_sum<<<blocks, B, 0, stream>>>(src, dst, el, er, m, sbuf, EH);
  }

  {
    long long EHF = E * HF;
    edge_agg<<<(int)((EHF + B - 1) / B), B, 0, stream>>>(src, dst, el, er, m, sbuf,
                                                         feat, rst, EHF);
  }

  {
    long long n = (long long)N * FGAT;
    finalize<<<(int)((n + B - 1) / B), B, 0, stream>>>(rst, bias, out, N);
  }
}
